// TverskyProjection_68496138437290
// MI455X (gfx1250) — matrix operand order repaired and checked by reading, not compiled, not run on hardware
//
#include <hip/hip_runtime.h>
#include <hip/hip_bf16.h>

typedef __attribute__((ext_vector_type(16))) _Float16 v16h;
typedef __attribute__((ext_vector_type(8)))  _Float16 v8h;
typedef __attribute__((ext_vector_type(8)))  float    v8f;

#define N_ROWS 16384
#define IN_F   1024
#define OUT_F  4096
#define NUM_F  16
#define K2     32   // width of fused X' / P' factors

// fast sigmoid(5v): v_exp_f32 + v_rcp_f32 (no IEEE div expansion)
__device__ __forceinline__ float sigm5(float v) {
    return __builtin_amdgcn_rcpf(1.0f + __expf(-5.0f * v));
}

__device__ __forceinline__ float bf16r(float x) {
    return __bfloat162float(__float2bfloat16(x));
}

// ---------------------------------------------------------------------------
// Phase 1: ternary-quantize prototypes (bf16 group-64), project onto features,
// fuse gating + theta/alpha/beta into P'[4096][32] (f16).
// One wave per prototype row; features cached in LDS (64 KB).
// ---------------------------------------------------------------------------
__global__ void __launch_bounds__(256) proto_kernel(
    const float* __restrict__ proto, const float* __restrict__ feat,
    const float* __restrict__ pTheta, const float* __restrict__ pAlpha,
    const float* __restrict__ pBeta, _Float16* __restrict__ Pp)
{
    __shared__ float sfeat[NUM_F * IN_F];   // 64 KB of the 320 KB WGP LDS
    for (int i = threadIdx.x; i < NUM_F * IN_F; i += 256) sfeat[i] = feat[i];
    __syncthreads();

    const float th = fabsf(pTheta[0]);
    const float al = fabsf(pAlpha[0]);
    const float be = fabsf(pBeta[0]);

    const int wave = threadIdx.x >> 5;
    const int lane = threadIdx.x & 31;
    const int row  = blockIdx.x * 8 + wave;
    const float* w = proto + (size_t)row * IN_F;

    float acc[NUM_F];
#pragma unroll
    for (int k = 0; k < NUM_F; ++k) acc[k] = 0.0f;

    // 16 groups of 64 contiguous weights per row; lane l covers g*64+l, g*64+32+l
    for (int g = 0; g < IN_F / 64; ++g) {
        const int i0 = g * 64 + lane;
        const int i1 = i0 + 32;
        const float w0 = bf16r(w[i0]);
        const float w1 = bf16r(w[i1]);
        float s = fabsf(w0) + fabsf(w1);
#pragma unroll
        for (int m = 16; m >= 1; m >>= 1) s += __shfl_xor(s, m, 32);
        const float scale = fmaxf(bf16r(s * (1.0f / 64.0f)), 1e-8f);
        const float rs = __builtin_amdgcn_rcpf(scale);  // feeds rint+clamp only
        const float q0 = bf16r(fminf(fmaxf(rintf(w0 * rs), -1.0f), 1.0f) * scale);
        const float q1 = bf16r(fminf(fmaxf(rintf(w1 * rs), -1.0f), 1.0f) * scale);
#pragma unroll
        for (int k = 0; k < NUM_F; ++k) {
            acc[k] = fmaf(q0, sfeat[k * IN_F + i0], acc[k]);
            acc[k] = fmaf(q1, sfeat[k * IN_F + i1], acc[k]);
        }
    }
    // butterfly reduce the 16 partial dots across the wave (wave32)
#pragma unroll
    for (int k = 0; k < NUM_F; ++k) {
#pragma unroll
        for (int m = 16; m >= 1; m >>= 1) acc[k] += __shfl_xor(acc[k], m, 32);
    }
    if (lane == 0) {
        _Float16* dst = Pp + (size_t)row * K2;
#pragma unroll
        for (int k = 0; k < NUM_F; ++k) {
            const float pf = acc[k];
            const float ps = sigm5(pf);
            const float pa = pf * ps;
            dst[k]         = (_Float16)(th * pa - al * (1.0f - ps)); // pairs with x_a
            dst[NUM_F + k] = (_Float16)(-be * pa);                   // pairs with (1-x_s)
        }
    }
}

// ---------------------------------------------------------------------------
// Phase 2: x_f = x @ features.T via WMMA; one wave per 16-row tile, K loop of
// 32 over 1024. Produces gated X'[16384][32] (f16).
// A (16x32 f16) per-lane layout: K = 16*half+{0..15}.
// B (32x16 f16) per-lane layout: n = lane&15, K = 16*half+{0..15}.
// ---------------------------------------------------------------------------
__global__ void __launch_bounds__(256) xproj_kernel(
    const float* __restrict__ x, const float* __restrict__ feat,
    _Float16* __restrict__ Xp)
{
    const int wave  = threadIdx.x >> 5;
    const int lane  = threadIdx.x & 31;
    const int half  = lane >> 4;
    const int r     = lane & 15;
    const int tileM = blockIdx.x * 8 + wave;

    const float* xrow = x + (size_t)(tileM * 16 + r) * IN_F;
    const float* frow = feat + (size_t)r * IN_F;   // B[k][n]=features[n][k], n=r

    v8f c = {};
    for (int kk = 0; kk < IN_F; kk += 32) {
        const float* a0 = xrow + kk + 16 * half;
        const float* a1 = xrow + kk + 16 * half + 8;
        v16h a;
#pragma unroll
        for (int e = 0; e < 8; ++e) { a[e] = (_Float16)a0[e]; a[8 + e] = (_Float16)a1[e]; }
        const float* b0 = frow + kk + 16 * half;
        v16h b;
#pragma unroll
        for (int e = 0; e < 16; ++e) b[e] = (_Float16)b0[e];
        c = __builtin_amdgcn_wmma_f32_16x16x32_f16(false, a, false, b, (short)0, c,
                                                   false, false);
    }
    // C layout: VGPR j -> M = j + 8*half (lanes split), N = lane&15 = feature k
#pragma unroll
    for (int j = 0; j < 8; ++j) {
        const int M = tileM * 16 + j + 8 * half;
        const float v = c[j];
        const float s = sigm5(v);
        _Float16* dst = Xp + (size_t)M * K2;
        dst[r]      = (_Float16)(v * s);      // x_a
        dst[16 + r] = (_Float16)(1.0f - s);   // 1 - x_s
    }
}

// ---------------------------------------------------------------------------
// Phase 3: out = X' @ P'.T, one v_wmma_f32_16x16x32_f16 per 16x16 tile.
// Wave loads A once, sweeps 8 column tiles (B = 32B contiguous per lane).
// Output writes (256 MB) are the bandwidth floor (~11.5 us at 23.3 TB/s).
// ---------------------------------------------------------------------------
__global__ void __launch_bounds__(256) out_kernel(
    const _Float16* __restrict__ Xp, const _Float16* __restrict__ Pp,
    float* __restrict__ out)
{
    const int wave   = threadIdx.x >> 5;
    const int lane   = threadIdx.x & 31;
    const int half   = lane >> 4;
    const int r      = lane & 15;
    const int task   = blockIdx.x * 8 + wave;  // 1024 M-tiles x 32 N-groups
    const int tileM  = task >> 5;
    const int groupN = task & 31;

    const _Float16* arow = Xp + (size_t)(tileM * 16 + r) * K2;
    const v8h a0 = *(const v8h*)(arow + 16 * half);
    const v8h a1 = *(const v8h*)(arow + 16 * half + 8);
    v16h a;
#pragma unroll
    for (int e = 0; e < 8; ++e) { a[e] = a0[e]; a[8 + e] = a1[e]; }

    float* obase = out + (size_t)(tileM * 16 + 8 * half) * OUT_F;

#pragma unroll
    for (int t = 0; t < 8; ++t) {
        const int n0 = (groupN * 8 + t) * 16;
        const v16h b = *(const v16h*)(Pp + (size_t)(n0 + r) * K2 + 16 * half);
        v8f c = {};
        c = __builtin_amdgcn_wmma_f32_16x16x32_f16(false, a, false, b, (short)0, c,
                                                   false, false);
#pragma unroll
        for (int j = 0; j < 8; ++j) {
            obase[(size_t)j * OUT_F + n0 + r] = c[j];
        }
    }
}

// ---------------------------------------------------------------------------
extern "C" void kernel_launch(void* const* d_in, const int* in_sizes, int n_in,
                              void* d_out, int out_size, void* d_ws, size_t ws_size,
                              hipStream_t stream) {
    const float* x     = (const float*)d_in[0];
    const float* feat  = (const float*)d_in[1];
    const float* proto = (const float*)d_in[2];
    const float* th    = (const float*)d_in[3];
    const float* al    = (const float*)d_in[4];
    const float* be    = (const float*)d_in[5];
    float* out = (float*)d_out;

    _Float16* Pp = (_Float16*)d_ws;                    // 4096*32*2  = 256 KB
    _Float16* Xp = Pp + (size_t)OUT_F * K2;            // 16384*32*2 = 1 MB

    proto_kernel<<<OUT_F / 8, 256, 0, stream>>>(proto, feat, th, al, be, Pp);
    xproj_kernel<<<(N_ROWS / 16) / 8, 256, 0, stream>>>(x, feat, Xp);
    out_kernel<<<(N_ROWS / 16) * (OUT_F / 16) / 64, 256, 0, stream>>>(Xp, Pp, out);
}
